// lstm_crf_32547262169403
// MI455X (gfx1250) — matrix operand order repaired and checked by reading, not compiled, not run on hardware
//
#include <hip/hip_runtime.h>
#include <hip/hip_bf16.h>

// ---------------------------------------------------------------------------
// BiLSTM-CRF NLL for MI455X (gfx1250), wave32 + WMMA f16 (f32 accumulate).
// Dims: B=64, T=512, V=128000, E=50, H=150, 4H=600, HID=300, L=2, K=48
// Padded: E->64, H->160, HID->320, 4H->640 (each gate 150->160)
// Recurrent kernel: Wh resident in LDS (205KB) + h ping-pong (41KB) = 240KB,
// c state in wave registers, G slice for t+1 prefetched each step.
// ---------------------------------------------------------------------------

typedef __attribute__((ext_vector_type(16))) _Float16 v16h;
typedef __attribute__((ext_vector_type(8)))  float    v8f;

#define BB   64
#define TT   512
#define EE   50
#define HH   150
#define KK   48
#define EP   64      // padded E
#define HP   160     // padded H
#define HIDP 320     // padded HID
#define NP   640     // padded 4H (4 gates x 160)
#define MROWS (TT*BB)   // 32768 time-major rows

// ---- WMMA helpers ----------------------------------------------------------

__device__ __forceinline__ v8f wmma_f32(v16h a, v16h b, v8f c) {
  return __builtin_amdgcn_wmma_f32_16x16x32_f16(
      /*neg_a=*/false, a, /*neg_b=*/false, b,
      /*c_mod=*/(short)0, c, /*reuse_a=*/false, /*reuse_b=*/false);
}

// A-matrix fragment, 16x32 (MxK), row-major source with leading dim ld.
// ISA layout: lanes 0-15 -> M rows, K pairs; lane-half selects K 0-7/8-15 (+16 grp).
__device__ __forceinline__ v16h load_a_frag(const _Float16* p, int ld) {
  const int lane = threadIdx.x & 31;
  const _Float16* row = p + (lane & 15) * ld;
  const int kh = (lane >> 4) * 8;
  v16h f;
#pragma unroll
  for (int g = 0; g < 2; ++g) {
#pragma unroll
    for (int vi = 0; vi < 4; ++vi) {
      const int k = g * 16 + kh + vi * 2;
      f[g * 8 + vi * 2]     = row[k];
      f[g * 8 + vi * 2 + 1] = row[k + 1];
    }
  }
  return f;
}

// B-matrix fragment, 32x16 (KxN). Source is the weight stored [N, K] row-major
// (i.e. W[n,k], computing A@W^T). Lanes 0-15 -> columns, lane-half selects K 0-7/8-15 (+16 grp).
__device__ __forceinline__ v16h load_b_frag(const _Float16* p, int ld) {
  const int lane = threadIdx.x & 31;
  const _Float16* row = p + (lane & 15) * ld + (lane >> 4) * 8;
  v16h f;
#pragma unroll
  for (int k = 0; k < 8; ++k) { f[k] = row[k]; f[8 + k] = row[16 + k]; }
  return f;
}

__device__ __forceinline__ float sigf(float x) { return 1.0f / (1.0f + expf(-x)); }

// ---- weight/bias prep ------------------------------------------------------
// rowGatePad: dst row r -> gate q=r/160, j=r%160, src row q*150+j (j<150)
// colGap:     dst col c -> src c (c<150) or c-10 (160<=c<160+(scols-150))
__global__ void pad_weight_kernel(const float* __restrict__ src, int srows, int scols,
                                  _Float16* __restrict__ dst, int drows, int dcols,
                                  int rowGatePad, int colGap) {
  int idx = blockIdx.x * blockDim.x + threadIdx.x;
  int total = drows * dcols;
  if (idx >= total) return;
  int r = idx / dcols, c = idx % dcols;
  int srow, scol;
  if (rowGatePad) { int q = r / 160, j = r % 160; srow = (j < 150) ? q * 150 + j : -1; }
  else            { srow = (r < srows) ? r : -1; }
  if (colGap) {
    if (c < 150) scol = c;
    else if (c >= 160 && (c - 10) < scols) scol = c - 10;
    else scol = -1;
  } else {
    scol = (c < scols) ? c : -1;
  }
  float v = (srow >= 0 && scol >= 0) ? src[(size_t)srow * scols + scol] : 0.0f;
  dst[idx] = (_Float16)v;
}

__global__ void pad_bias_kernel(const float* __restrict__ bi, const float* __restrict__ bh,
                                float* __restrict__ dst) {
  int r = blockIdx.x * blockDim.x + threadIdx.x;
  if (r >= NP) return;
  int q = r / 160, j = r % 160;
  dst[r] = (j < 150) ? bi[q * 150 + j] + bh[q * 150 + j] : 0.0f;
}

// ---- embedding gather to time-major f16 [T*B, 64] --------------------------
__global__ void embed_kernel(const int* __restrict__ x, const float* __restrict__ embed,
                             _Float16* __restrict__ xe) {
  int gid = blockIdx.x * blockDim.x + threadIdx.x;
  if (gid >= MROWS * EP) return;
  int rid = gid >> 6;           // t*64 + b
  int e   = gid & 63;
  int t = rid >> 6, b = rid & 63;
  int tok = x[b * TT + t];
  float v = (e < EE) ? embed[(size_t)tok * EE + e] : 0.0f;
  xe[(size_t)rid * EP + e] = (_Float16)v;
}

// ---- input projection GEMM: G[32768, 640] = A[32768, lda] @ W[640, lda]^T + bias
__global__ void __launch_bounds__(256)
input_gemm_kernel(const _Float16* __restrict__ A, int lda, int ksteps,
                  const _Float16* __restrict__ W, const float* __restrict__ bias,
                  float* __restrict__ G) {
  int wave = (blockIdx.x * blockDim.x + threadIdx.x) >> 5;   // one 16x16 tile / wave
  int mt = wave / (NP / 16), nt = wave % (NP / 16);
  int m0 = mt * 16, n0 = nt * 16;
  v8f acc = {};
  for (int ks = 0; ks < ksteps; ++ks) {
    v16h a = load_a_frag(A + (size_t)m0 * lda + ks * 32, lda);
    v16h b = load_b_frag(W + (size_t)n0 * lda + ks * 32, lda);
    acc = wmma_f32(a, b, acc);
  }
  int lane = threadIdx.x & 31;
  int n = n0 + (lane & 15);
  int mb = m0 + 8 * (lane >> 4);
  float bv = bias[n];
#pragma unroll
  for (int r = 0; r < 8; ++r)
    G[(size_t)(mb + r) * NP + n] = acc[r] + bv;
}

// ---- recurrent LSTM: grid.x = 2 (0=fwd, 1=bwd), one workgroup per direction.
// Wh lives in LDS (loaded once); h ping-pongs in LDS; c lives in wave registers.
// 8 waves x 5 tasks cover 4 (M) x 10 (j) tiles; each task does 4 gate tiles.
__global__ void __launch_bounds__(256)
lstm_rec_kernel(const float* __restrict__ G0, const float* __restrict__ G1,
                const _Float16* __restrict__ Whf, const _Float16* __restrict__ Whb,
                _Float16* __restrict__ hout) {
  __shared__ _Float16 hbuf[2][BB * HP];   // 2 x 20KB
  __shared__ _Float16 WhS[NP * HP];       // 200KB, resident all 512 steps
  const int dir = blockIdx.x;
  const float*     G  = dir ? G1  : G0;
  const _Float16*  Wh = dir ? Whb : Whf;
  const int col_off = dir ? HP : 0;

  // one-time: zero h state, stage Wh into LDS (4B words)
  for (int i = threadIdx.x; i < BB * HP; i += blockDim.x) hbuf[0][i] = (_Float16)0;
  {
    const unsigned int* src = (const unsigned int*)Wh;
    unsigned int* dst = (unsigned int*)WhS;
    for (int i = threadIdx.x; i < NP * HP / 2; i += blockDim.x) dst[i] = src[i];
  }
  __syncthreads();

  const int wv   = threadIdx.x >> 5;
  const int lane = threadIdx.x & 31;
  const int nloc = lane & 15;
  const int mhi  = (lane >> 4) * 8;

  v8f zero = {};
  v8f cfrag[5];
#pragma unroll
  for (int i = 0; i < 5; ++i) cfrag[i] = zero;

  for (int step = 0; step < TT; ++step) {
    const int t   = dir ? (TT - 1 - step) : step;
    const int cur = step & 1, nxt = cur ^ 1;

    // prefetch next step's gate pre-activations (164KB slice) to overlap WMMAs
    if (step + 1 < TT) {
      const int tn = dir ? (t - 1) : (t + 1);
      const float* pf = G + (size_t)tn * BB * NP;
      for (int i = threadIdx.x * 64; i < BB * NP; i += 256 * 64)
        __builtin_prefetch(pf + i, 0, 0);
    }

#pragma unroll
    for (int i = 0; i < 5; ++i) {
      const int task = wv + 8 * i;                 // 0..39
      const int m0 = (task / 10) * 16;             // batch tile
      const int j0 = (task % 10) * 16;             // within-gate column tile
      v8f acc0 = zero, acc1 = zero, acc2 = zero, acc3 = zero;
#pragma unroll
      for (int ks = 0; ks < 5; ++ks) {             // K = 160
        v16h a = load_a_frag(&hbuf[cur][m0 * HP + ks * 32], HP);
        v16h b0 = load_b_frag(&WhS[(0 * 160 + j0) * HP + ks * 32], HP);
        acc0 = wmma_f32(a, b0, acc0);
        v16h b1 = load_b_frag(&WhS[(1 * 160 + j0) * HP + ks * 32], HP);
        acc1 = wmma_f32(a, b1, acc1);
        v16h b2 = load_b_frag(&WhS[(2 * 160 + j0) * HP + ks * 32], HP);
        acc2 = wmma_f32(a, b2, acc2);
        v16h b3 = load_b_frag(&WhS[(3 * 160 + j0) * HP + ks * 32], HP);
        acc3 = wmma_f32(a, b3, acc3);
      }
      const float* gp = G + (size_t)t * BB * NP;
#pragma unroll
      for (int r = 0; r < 8; ++r) {
        const int m = m0 + mhi + r;
        const int j = j0 + nloc;
        const float* g = gp + (size_t)m * NP + j;
        float gi = acc0[r] + g[0];
        float gf = acc1[r] + g[160];
        float gg = acc2[r] + g[320];
        float go = acc3[r] + g[480];
        float c = sigf(gf) * cfrag[i][r] + sigf(gi) * tanhf(gg);
        cfrag[i][r] = c;
        float h = sigf(go) * tanhf(c);
        _Float16 hh = (j < HH) ? (_Float16)h : (_Float16)0;
        hbuf[nxt][m * HP + j] = hh;
        hout[((size_t)t * BB + m) * HIDP + col_off + j] = hh;
      }
    }
    __syncthreads();
  }
}

// ---- output projection -> masked emissions [B, T, K] f32 -------------------
__global__ void __launch_bounds__(256)
out_proj_kernel(const _Float16* __restrict__ A,      // h2 [32768, 320]
                const _Float16* __restrict__ W,      // outW padded [48, 320]
                const float* __restrict__ outb,
                const int* __restrict__ x,
                float* __restrict__ emit) {
  int wave = (blockIdx.x * blockDim.x + threadIdx.x) >> 5;
  int mt = wave / 3, nt = wave % 3;                  // 2048 x 3 tiles
  int m0 = mt * 16, n0 = nt * 16;
  v8f acc = {};
  for (int ks = 0; ks < HIDP / 32; ++ks) {
    v16h a = load_a_frag(A + (size_t)m0 * HIDP + ks * 32, HIDP);
    v16h b = load_b_frag(W + (size_t)n0 * HIDP + ks * 32, HIDP);
    acc = wmma_f32(a, b, acc);
  }
  int lane = threadIdx.x & 31;
  int n = n0 + (lane & 15);
  float bv = outb[n];
#pragma unroll
  for (int r = 0; r < 8; ++r) {
    int m = m0 + 8 * (lane >> 4) + r;                // m = t*64 + b
    int b = m & 63, t = m >> 6;
    float mask = (x[b * TT + t] > 0) ? 1.0f : 0.0f;
    emit[((size_t)b * TT + t) * KK + n] = (acc[r] + bv) * mask;
  }
}

// ---- CRF forward (partition) -----------------------------------------------
__global__ void crf_forward_kernel(const float* __restrict__ emit,
                                   const int* __restrict__ x,
                                   const float* __restrict__ trans,
                                   float* __restrict__ logZ) {
  __shared__ float Zs[KK];
  __shared__ float Ts[KK * KK];
  int b = blockIdx.x, f = threadIdx.x;
  for (int i = f; i < KK * KK; i += blockDim.x) Ts[i] = trans[i];
  if (f < KK) Zs[f] = (f == 2) ? 0.0f : -10000.0f;   // SOS = 2
  __syncthreads();
  for (int t = 0; t < TT; ++t) {
    float zn = 0.0f;
    if (f < KK) {
      float m = -3.0e38f;
      for (int k = 0; k < KK; ++k) { float v = Zs[k] + Ts[f * KK + k]; m = v > m ? v : m; }
      float s = 0.0f;
      for (int k = 0; k < KK; ++k) s += expf(Zs[k] + Ts[f * KK + k] - m);
      zn = emit[((size_t)b * TT + t) * KK + f] + m + logf(s);
    }
    int mt = x[b * TT + t] > 0;
    __syncthreads();
    if (f < KK && mt) Zs[f] = zn;
    __syncthreads();
  }
  if (f == 0) {
    float m = -3.0e38f;
    for (int k = 0; k < KK; ++k) m = Zs[k] > m ? Zs[k] : m;
    float s = 0.0f;
    for (int k = 0; k < KK; ++k) s += expf(Zs[k] - m);
    logZ[b] = m + logf(s);
  }
}

// ---- gold-path score + final NLL -------------------------------------------
__global__ void score_final_kernel(const float* __restrict__ emit,
                                   const int* __restrict__ x,
                                   const int* __restrict__ y0,
                                   const float* __restrict__ trans,
                                   const float* __restrict__ logZ,
                                   float* __restrict__ out) {
  int b = threadIdx.x;
  if (b >= BB) return;
  int yp = 2;                                        // SOS
  float sc = 0.0f;
  for (int t = 0; t < TT; ++t) {
    int yt = y0[b * TT + t];
    float mt = (x[b * TT + t] > 0) ? 1.0f : 0.0f;
    sc += emit[((size_t)b * TT + t) * KK + yt] + trans[yt * KK + yp] * mt;
    yp = yt;
  }
  out[b] = logZ[b] - sc;
}

// ---------------------------------------------------------------------------

extern "C" void kernel_launch(void* const* d_in, const int* in_sizes, int n_in,
                              void* d_out, int out_size, void* d_ws, size_t ws_size,
                              hipStream_t stream) {
  (void)in_sizes; (void)n_in; (void)out_size; (void)ws_size;

  const int*   x     = (const int*)d_in[0];
  const int*   y0    = (const int*)d_in[1];
  const float* embed = (const float*)d_in[2];
  // lstm_params flattened: l0_fwd, l0_bwd, l1_fwd, l1_bwd; each Wi, Wh, bi, bh
  const float* Wi[4], *Wh[4], *bi[4], *bh[4];
  for (int p = 0; p < 4; ++p) {
    Wi[p] = (const float*)d_in[3 + 4 * p + 0];
    Wh[p] = (const float*)d_in[3 + 4 * p + 1];
    bi[p] = (const float*)d_in[3 + 4 * p + 2];
    bh[p] = (const float*)d_in[3 + 4 * p + 3];
  }
  const float* outW  = (const float*)d_in[19];
  const float* outb  = (const float*)d_in[20];
  const float* trans = (const float*)d_in[21];
  float* out = (float*)d_out;

  // workspace carve-out
  char* base = (char*)d_ws;
  size_t off = 0;
  auto carve = [&](size_t bytes) -> char* {
    off = (off + 255) & ~(size_t)255;
    char* p = base + off;
    off += bytes;
    return p;
  };
  _Float16* xe  = (_Float16*)carve((size_t)MROWS * EP * 2);
  _Float16* h1  = (_Float16*)carve((size_t)MROWS * HIDP * 2);
  _Float16* h2  = (_Float16*)carve((size_t)MROWS * HIDP * 2);
  float*    G0  = (float*)carve((size_t)MROWS * NP * 4);
  float*    G1  = (float*)carve((size_t)MROWS * NP * 4);
  _Float16* wiP[4]; _Float16* whP[4]; float* bP[4];
  for (int p = 0; p < 4; ++p) {
    int din = (p < 2) ? EP : HIDP;
    wiP[p] = (_Float16*)carve((size_t)NP * din * 2);
    whP[p] = (_Float16*)carve((size_t)NP * HP * 2);
    bP[p]  = (float*)carve((size_t)NP * 4);
  }
  _Float16* owP  = (_Float16*)carve((size_t)KK * HIDP * 2);
  float*    emit = (float*)carve((size_t)BB * TT * KK * 4);
  float*    logZ = (float*)carve((size_t)BB * 4);

  auto blocks = [](size_t total, int bdim) { return (unsigned)((total + bdim - 1) / bdim); };

  // 1) weight/bias prep
  for (int p = 0; p < 4; ++p) {
    int din  = (p < 2) ? EE : HH * 2;      // 50 or 300 source cols
    int dinP = (p < 2) ? EP : HIDP;
    int cgap = (p < 2) ? 0 : 1;            // layer-1 input has 150..299 -> gap at 150
    pad_weight_kernel<<<blocks((size_t)NP * dinP, 256), 256, 0, stream>>>(
        Wi[p], 600, din, wiP[p], NP, dinP, /*rowGatePad=*/1, cgap);
    pad_weight_kernel<<<blocks((size_t)NP * HP, 256), 256, 0, stream>>>(
        Wh[p], 600, HH, whP[p], NP, HP, /*rowGatePad=*/1, /*colGap=*/0);
    pad_bias_kernel<<<blocks(NP, 256), 256, 0, stream>>>(bi[p], bh[p], bP[p]);
  }
  pad_weight_kernel<<<blocks((size_t)KK * HIDP, 256), 256, 0, stream>>>(
      outW, KK, 300, owP, KK, HIDP, /*rowGatePad=*/0, /*colGap=*/1);

  // 2) embedding gather (time-major f16)
  embed_kernel<<<blocks((size_t)MROWS * EP, 256), 256, 0, stream>>>(x, embed, xe);

  // 3) layer 0: input GEMMs then recurrence (fwd+bwd concurrent)
  const unsigned gemm_blocks = (unsigned)((size_t)(MROWS / 16) * (NP / 16) / 8); // 10240
  input_gemm_kernel<<<gemm_blocks, 256, 0, stream>>>(xe, EP, EP / 32, wiP[0], bP[0], G0);
  input_gemm_kernel<<<gemm_blocks, 256, 0, stream>>>(xe, EP, EP / 32, wiP[1], bP[1], G1);
  lstm_rec_kernel<<<2, 256, 0, stream>>>(G0, G1, whP[0], whP[1], h1);

  // 4) layer 1
  input_gemm_kernel<<<gemm_blocks, 256, 0, stream>>>(h1, HIDP, HIDP / 32, wiP[2], bP[2], G0);
  input_gemm_kernel<<<gemm_blocks, 256, 0, stream>>>(h1, HIDP, HIDP / 32, wiP[3], bP[3], G1);
  lstm_rec_kernel<<<2, 256, 0, stream>>>(G0, G1, whP[2], whP[3], h2);

  // 5) output projection -> masked emissions
  const unsigned proj_blocks = (unsigned)((size_t)(MROWS / 16) * 3 / 8);         // 768
  out_proj_kernel<<<proj_blocks, 256, 0, stream>>>(h2, owP, outb, x, emit);

  // 6) CRF partition + gold score -> NLL
  crf_forward_kernel<<<BB, 64, 0, stream>>>(emit, x, trans, logZ);
  score_final_kernel<<<1, 64, 0, stream>>>(emit, x, y0, trans, logZ, out);
}
